// Encoder_44135083934178
// MI455X (gfx1250) — matrix operand order repaired and checked by reading, not compiled, not run on hardware
//
#include <hip/hip_runtime.h>
#include <cstddef>

// ---------------- CDNA5 types ----------------
typedef __attribute__((ext_vector_type(16))) __bf16 v16bf;
typedef __attribute__((ext_vector_type(8)))  __bf16 v8bf;
typedef __attribute__((ext_vector_type(8)))  float  v8f;
typedef unsigned short u16;

// ---------------- problem constants ----------------
#define BATCH    4
#define TSEQ     2048
#define T2LEN    2050
#define DMODEL   512
#define NHEADS   8
#define DHEAD    64
#define DFF      2048
#define NLAYERS  4
#define WINR     15
#define WINSZ    31
#define KME1     261   // 2*128 + 5
#define KPAD     320   // KME1 rounded up to BK (64)

// ---------------- GEMM tiling ----------------
#define BM 128
#define BN 128
#define BK 64
#define LDA 72   // padded bf16 row stride for A tile (rows 144B, 16B aligned)
#define LDB 72   // padded bf16 row stride for B tile (stored [n][k])
#define ACH ((BM * BK) / (8 * 256))   // 4: A 16B-chunks per thread
#define BCH ((BN * BK) / (8 * 256))   // 4: B 16B-chunks per thread

// fp32 -> bf16, round-half-up (1 VALU)
__device__ __forceinline__ u16 bfh(float f) {
    return (u16)((__float_as_uint(f) + 0x8000u) >> 16);
}
// two fp32 -> packed 2xbf16 dword: 2 adds + 1 v_perm_b32
__device__ __forceinline__ unsigned pkbf(float a, float b) {
#if __has_builtin(__builtin_amdgcn_perm)
    return __builtin_amdgcn_perm(__float_as_uint(b) + 0x8000u,
                                 __float_as_uint(a) + 0x8000u, 0x07060302u);
#else
    return (unsigned)bfh(a) | ((unsigned)bfh(b) << 16);
#endif
}

// C[M,N] = epilogue(A[M,K] @ Wt^T + bias).  A: bf16 [M][K] (K%64==0, padded).
// Wt: bf16 [N][K] pre-transposed, pre-converted.  N%128==0, M arbitrary.
// ACT: 0=none, 1=PReLU(alpha). RES: C += r (fp32). OUTBF: write bf16 to Cb.
template<int ACT, bool RES, bool OUTBF>
__global__ __launch_bounds__(256)
void gemm_kernel(const u16* __restrict__ A, const u16* __restrict__ Wt,
                 const float* __restrict__ bias, const float* __restrict__ alpha_p,
                 float* __restrict__ C, u16* __restrict__ Cb,
                 int M, int N, int K)
{
    __shared__ u16 As[BM * LDA];
    __shared__ u16 Bs[BN * LDB];

    const int tid  = threadIdx.x;
    const int lane = tid & 31;
    const int wave = tid >> 5;
    const int wm   = wave >> 1;        // 0..3  (M direction, 32 rows each)
    const int wn   = wave & 1;         // 0..1  (N direction, 64 cols each)
    const int bm0  = blockIdx.y * BM;
    const int bn0  = blockIdx.x * BN;
    const int halo = lane >> 4;        // 0 | 1 (lane half)
    const int l16  = lane & 15;

    v8f acc[2][4] = {};

    // per-thread staging coordinates (k-invariant parts hoisted)
    int ar[ACH], ac[ACH], bn_[BCH], bc[BCH];
    #pragma unroll
    for (int i = 0; i < ACH; ++i) {
        int e = tid + i * 256;
        int r = e >> 3;                              // 8 chunks per A row
        ac[i] = (e & 7) * 8;
        int gr = bm0 + r; if (gr >= M) gr = M - 1;   // clamp, no predication
        ar[i] = gr;
    }
    #pragma unroll
    for (int i = 0; i < BCH; ++i) {
        int e = tid + i * 256;
        bn_[i] = e >> 3;
        bc[i]  = (e & 7) * 8;
    }

    for (int k0 = 0; k0 < K; k0 += BK) {
        // ---- phase 1: issue ALL staging loads (keeps 8 loads in flight) ----
        uint4 ta[ACH], tw[BCH];
        #pragma unroll
        for (int i = 0; i < ACH; ++i)
            ta[i] = *reinterpret_cast<const uint4*>(&A[(size_t)ar[i] * K + k0 + ac[i]]);
        #pragma unroll
        for (int i = 0; i < BCH; ++i)
            tw[i] = *reinterpret_cast<const uint4*>(&Wt[(size_t)(bn0 + bn_[i]) * K + k0 + bc[i]]);
        // ---- phase 2: store to LDS (partial loadcnt waits, no serialization) ----
        #pragma unroll
        for (int i = 0; i < ACH; ++i) {
            int r = (tid + i * 256) >> 3;
            *reinterpret_cast<uint4*>(&As[r * LDA + ac[i]]) = ta[i];
        }
        #pragma unroll
        for (int i = 0; i < BCH; ++i)
            *reinterpret_cast<uint4*>(&Bs[bn_[i] * LDB + bc[i]]) = tw[i];
        __syncthreads();

        // two WMMA k-steps per staged tile
        #pragma unroll
        for (int ks = 0; ks < 2; ++ks) {
            // A 16x32 layout: lanes 0-15 hold K 0-7 & 16-23, lanes 16-31 hold 8-15 & 24-31
            v16bf afr[2], bfr[4];
            #pragma unroll
            for (int mi = 0; mi < 2; ++mi) {
                int row = wm * 32 + mi * 16 + l16;
                int kb  = ks * 32 + halo * 8;
                v8bf lo = *reinterpret_cast<const v8bf*>(&As[row * LDA + kb]);
                v8bf hi = *reinterpret_cast<const v8bf*>(&As[row * LDA + kb + 16]);
                afr[mi] = __builtin_shufflevector(lo, hi, 0,1,2,3,4,5,6,7,8,9,10,11,12,13,14,15);
            }
            // B 32x16 layout: lanes 0-15 hold K 0-7 & 16-23, lanes 16-31 hold 8-15 & 24-31 (Bs is [n][k])
            #pragma unroll
            for (int ni = 0; ni < 4; ++ni) {
                int col = wn * 64 + ni * 16 + l16;
                int kb  = ks * 32 + halo * 8;
                v8bf lo = *reinterpret_cast<const v8bf*>(&Bs[col * LDB + kb]);
                v8bf hi = *reinterpret_cast<const v8bf*>(&Bs[col * LDB + kb + 16]);
                bfr[ni] = __builtin_shufflevector(lo, hi, 0,1,2,3,4,5,6,7,8,9,10,11,12,13,14,15);
            }
            #pragma unroll
            for (int mi = 0; mi < 2; ++mi)
                #pragma unroll
                for (int ni = 0; ni < 4; ++ni)
                    acc[mi][ni] = __builtin_amdgcn_wmma_f32_16x16x32_bf16(
                        false, afr[mi], false, bfr[ni], (short)0, acc[mi][ni], false, false);
        }
        __syncthreads();
    }

    const float al = (ACT == 1 && alpha_p) ? *alpha_p : 0.f;
    const bool full = (bm0 + BM <= M);           // uniform fast path
    // C/D layout: VGPR r -> M = r (lanes 0-15) or r+8 (lanes 16-31); N = lane&15
    #pragma unroll
    for (int mi = 0; mi < 2; ++mi)
        #pragma unroll
        for (int ni = 0; ni < 4; ++ni) {
            int col = bn0 + wn * 64 + ni * 16 + l16;
            float bcol = bias[col];
            #pragma unroll
            for (int r = 0; r < 8; ++r) {
                int row = bm0 + wm * 32 + mi * 16 + r + halo * 8;
                if (full || row < M) {
                    float v = acc[mi][ni][r] + bcol;
                    if (ACT == 1) v = (v >= 0.f) ? v : al * v;
                    size_t o = (size_t)row * N + col;
                    if (OUTBF) {
                        Cb[o] = bfh(v);
                    } else {
                        if (RES) v += C[o];
                        C[o] = v;
                    }
                }
            }
        }
}

// ---- weight transform: W[Kw][N] fp32 -> Wt[N][K] bf16, zero-pad k>=Kw ----
__global__ __launch_bounds__(256)
void wtrans_kernel(const float* __restrict__ W, u16* __restrict__ Wt,
                   int K, int N, int Kw)
{
    __shared__ float tile[32][33];
    int kt = blockIdx.x * 32;
    int nt = blockIdx.y * 32;
    int tx = threadIdx.x & 31;
    int ty = threadIdx.x >> 5;
    #pragma unroll
    for (int i = 0; i < 32; i += 8) {
        int kk = kt + ty + i;
        tile[ty + i][tx] = (kk < Kw) ? W[(size_t)kk * N + nt + tx] : 0.f;
    }
    __syncthreads();
    #pragma unroll
    for (int i = 0; i < 32; i += 8) {
        int nn = nt + ty + i;
        Wt[(size_t)nn * K + kt + tx] = bfh(tile[tx][ty + i]);
    }
}

// ---------------- LayerNorm: one wave per 512-wide row, bf16 out ----------------
__global__ __launch_bounds__(256)
void ln_kernel(const float* __restrict__ x, const float* __restrict__ g,
               const float* __restrict__ bb, u16* __restrict__ y, int rows)
{
    int lane = threadIdx.x & 31, wave = threadIdx.x >> 5;
    int row = blockIdx.x * 8 + wave;
    if (row >= rows) return;
    const float2* xr = reinterpret_cast<const float2*>(x + (size_t)row * DMODEL);
    float2 r[8];
    float sum = 0.f, sq = 0.f;
    #pragma unroll
    for (int i = 0; i < 8; ++i) {
        float2 v = xr[lane + i * 32];
        r[i] = v; sum += v.x + v.y; sq += v.x * v.x + v.y * v.y;
    }
    #pragma unroll
    for (int off = 16; off; off >>= 1) {
        sum += __shfl_xor(sum, off);
        sq  += __shfl_xor(sq,  off);
    }
    float mean = sum * (1.f / 512.f);
    float var  = sq * (1.f / 512.f) - mean * mean;
    float rs   = rsqrtf(var + 1e-5f);
    unsigned* yr = reinterpret_cast<unsigned*>(y + (size_t)row * DMODEL);
    #pragma unroll
    for (int i = 0; i < 8; ++i) {
        int p = lane + i * 32;           // pair index
        float a = (r[i].x - mean) * rs * g[2 * p]     + bb[2 * p];
        float b = (r[i].y - mean) * rs * g[2 * p + 1] + bb[2 * p + 1];
        yr[p] = pkbf(a, b);
    }
}

// ---------------- windowed attention: one wave per (b,t,h) query, bf16 out ----------------
__global__ __launch_bounds__(256)
void attn_kernel(const float* __restrict__ q, const float* __restrict__ k,
                 const float* __restrict__ v, const float* __restrict__ tbl,
                 u16* __restrict__ o)
{
    __shared__ float pw[8][32];
    const int lane = threadIdx.x & 31;
    const int wave = threadIdx.x >> 5;
    int gq  = blockIdx.x * 8 + wave;           // 0 .. BATCH*T2LEN*NHEADS-1
    int b   = gq / (T2LEN * NHEADS);
    int rem = gq % (T2LEN * NHEADS);
    int t   = rem / NHEADS;
    int h   = rem % NHEADS;
    const size_t qbase = ((size_t)(b * T2LEN + t) * NHEADS + h) * DHEAD;
    const float scale = 0.125f;                 // 1/sqrt(64)

    // phase 1: lane = window offset; sc = (q.k + q.tbl)*scale = q.(k+tbl)*scale
    float s = -1.0e9f;
    if (lane < WINSZ) {
        int idx = t + lane - WINR;
        bool valid = (idx >= 0) && (idx < T2LEN);
        int idxc = idx < 0 ? 0 : (idx > T2LEN - 1 ? T2LEN - 1 : idx);
        size_t kbase = ((size_t)(b * T2LEN + idxc) * NHEADS + h) * DHEAD;
        const float* tb = tbl + lane * DHEAD;
        float acc = 0.f;
        #pragma unroll 8
        for (int d = 0; d < DHEAD; ++d)
            acc += q[qbase + d] * (k[kbase + d] + tb[d]);
        s = valid ? acc * scale : -1.0e9f;
    }
    float mx = s;
    #pragma unroll
    for (int off = 16; off; off >>= 1) mx = fmaxf(mx, __shfl_xor(mx, off));
    float e = __expf(s - mx);
    if (lane >= WINSZ) e = 0.f;
    float sum = e;
    #pragma unroll
    for (int off = 16; off; off >>= 1) sum += __shfl_xor(sum, off);
    float p = e / sum;
    pw[wave][lane] = p;
    __builtin_amdgcn_wave_barrier();

    // phase 2: lane = pair of head dims; packed bf16 store
    int d0 = lane * 2;
    float a0 = 0.f, a1 = 0.f;
    for (int j = 0; j < WINSZ; ++j) {
        float pj = pw[wave][j];
        int idx = t + j - WINR;
        int idxc = idx < 0 ? 0 : (idx > T2LEN - 1 ? T2LEN - 1 : idx);
        size_t vb2 = ((size_t)(b * T2LEN + idxc) * NHEADS + h) * DHEAD + d0;
        a0 += pj * v[vb2];
        a1 += pj * v[vb2 + 1];
    }
    *reinterpret_cast<unsigned*>(o + qbase + d0) = pkbf(a0, a1);
}

// ---------------- small helper kernels ----------------
__global__ void concat_kernel(const float* __restrict__ x, const float* __restrict__ mask,
                              const float* __restrict__ traj, u16* __restrict__ Acat)
{
    int e = blockIdx.x * 256 + threadIdx.x;
    if (e >= BATCH * TSEQ * KPAD) return;
    int m = e / KPAD, c = e % KPAD;
    float v = 0.f;
    if (c < 128)       v = x[(size_t)m * 128 + c];
    else if (c < 256)  v = mask[(size_t)m * 128 + (c - 128)];
    else if (c < KME1) v = traj[(size_t)m * 5 + (c - 256)];
    Acat[e] = bfh(v);
}

__global__ void assemble_kernel(const float* __restrict__ henc, const float* __restrict__ mu,
                                const float* __restrict__ lv, float* __restrict__ h)
{
    int e = blockIdx.x * 256 + threadIdx.x;
    if (e >= BATCH * T2LEN * DMODEL) return;
    int row = e / DMODEL, d = e % DMODEL;
    int b = row / T2LEN, t = row % T2LEN;
    float v;
    if (t == 0)      v = mu[d];
    else if (t == 1) v = lv[d];
    else             v = henc[((size_t)(b * TSEQ + (t - 2))) * DMODEL + d];
    h[e] = v;
}

__global__ void relpos_kernel(const float* __restrict__ W1, const float* __restrict__ b1,
                              const float* __restrict__ a_p, const float* __restrict__ W2,
                              const float* __restrict__ b2, float* __restrict__ tbl)
{
    int e = blockIdx.x * 256 + threadIdx.x;
    if (e >= WINSZ * DHEAD) return;
    int o = e / DHEAD, d = e % DHEAD;
    float rel = (float)(o - WINR);
    float a = *a_p;
    float acc = b2[d];
    for (int j = 0; j < DMODEL; ++j) {
        float t = rel * W1[j] + b1[j];
        t = (t >= 0.f) ? t : a * t;
        acc += t * W2[(size_t)j * DHEAD + d];
    }
    tbl[e] = acc;
}

__global__ void out_kernel(const float* __restrict__ h, float* __restrict__ out)
{
    int e = blockIdx.x * 256 + threadIdx.x;
    if (e >= 2 * BATCH * DMODEL) return;
    int which = e / (BATCH * DMODEL);
    int rem   = e % (BATCH * DMODEL);
    int b = rem / DMODEL, d = rem % DMODEL;
    out[e] = h[((size_t)(b * T2LEN + which)) * DMODEL + d];
}

// ---------------- host launcher ----------------
extern "C" void kernel_launch(void* const* d_in, const int* in_sizes, int n_in,
                              void* d_out, int out_size, void* d_ws, size_t ws_size,
                              hipStream_t stream)
{
    (void)in_sizes; (void)n_in; (void)out_size; (void)ws_size;
    const float* x      = (const float*)d_in[0];
    const float* mask   = (const float*)d_in[1];
    const float* traj   = (const float*)d_in[2];
    const float* me_W1  = (const float*)d_in[3];
    const float* me_b1  = (const float*)d_in[4];
    const float* me_a1  = (const float*)d_in[5];
    const float* me_W2  = (const float*)d_in[6];
    const float* me_b2  = (const float*)d_in[7];
    const float* me_a2  = (const float*)d_in[8];
    const float* mu_tok = (const float*)d_in[9];
    const float* lv_tok = (const float*)d_in[10];
    const float* rp_W1  = (const float*)d_in[11];
    const float* rp_b1  = (const float*)d_in[12];
    const float* rp_a   = (const float*)d_in[13];
    const float* rp_W2  = (const float*)d_in[14];
    const float* rp_b2  = (const float*)d_in[15];
    const float* ln1_g  = (const float*)d_in[16];
    const float* ln1_b  = (const float*)d_in[17];
    const float* Wq     = (const float*)d_in[18];
    const float* bq     = (const float*)d_in[19];
    const float* Wk     = (const float*)d_in[20];
    const float* bk     = (const float*)d_in[21];
    const float* Wv     = (const float*)d_in[22];
    const float* bv     = (const float*)d_in[23];
    const float* Wo     = (const float*)d_in[24];
    const float* bo     = (const float*)d_in[25];
    const float* ln2_g  = (const float*)d_in[26];
    const float* ln2_b  = (const float*)d_in[27];
    const float* Wf1    = (const float*)d_in[28];
    const float* bf1    = (const float*)d_in[29];
    const float* af     = (const float*)d_in[30];
    const float* Wf2    = (const float*)d_in[31];
    const float* bf2    = (const float*)d_in[32];
    float* outf = (float*)d_out;

    // ---- workspace partition (≈160 MB) ----
    char* base = (char*)d_ws;
    size_t ofs = 0;
    auto allocF = [&](size_t n) { float* p = (float*)(base + ofs); ofs = (ofs + n * 4 + 255) & ~(size_t)255; return p; };
    auto allocH = [&](size_t n) { u16*   p = (u16*)  (base + ofs); ofs = (ofs + n * 2 + 255) & ~(size_t)255; return p; };

    const size_t ROWS = (size_t)BATCH * T2LEN;            // 8200
    float* h   = allocF(ROWS * DMODEL);
    float* qf  = allocF(ROWS * DMODEL);   // also ME2 fp32 out
    float* kf  = allocF(ROWS * DMODEL);
    float* vf  = allocF(ROWS * DMODEL);
    float* tb  = allocF(WINSZ * DHEAD);
    u16* acat  = allocH((size_t)BATCH * TSEQ * KPAD);
    u16* me1b  = allocH((size_t)BATCH * TSEQ * DMODEL);
    u16* hnb   = allocH(ROWS * DMODEL);
    u16* attb  = allocH(ROWS * DMODEL);
    u16* ff1b  = allocH(ROWS * DFF);
    u16* wtme1 = allocH((size_t)DMODEL * KPAD);
    u16* wtme2 = allocH((size_t)DMODEL * DMODEL);
    u16* wtq   = allocH((size_t)NLAYERS * DMODEL * DMODEL);
    u16* wtk   = allocH((size_t)NLAYERS * DMODEL * DMODEL);
    u16* wtv   = allocH((size_t)NLAYERS * DMODEL * DMODEL);
    u16* wto   = allocH((size_t)NLAYERS * DMODEL * DMODEL);
    u16* wtf1  = allocH((size_t)NLAYERS * DFF * DMODEL);
    u16* wtf2  = allocH((size_t)NLAYERS * DMODEL * DFF);

    dim3 blk(256);
    const int MROWS = (int)ROWS;

    // ---- one-time weight transforms (convert + transpose + pad) ----
    wtrans_kernel<<<dim3(KPAD / 32, DMODEL / 32), blk, 0, stream>>>(me_W1, wtme1, KPAD, DMODEL, KME1);
    wtrans_kernel<<<dim3(DMODEL / 32, DMODEL / 32), blk, 0, stream>>>(me_W2, wtme2, DMODEL, DMODEL, DMODEL);
    for (int i = 0; i < NLAYERS; ++i) {
        const size_t qo = (size_t)i * DMODEL * DMODEL;
        wtrans_kernel<<<dim3(DMODEL / 32, DMODEL / 32), blk, 0, stream>>>(Wq + qo, wtq + qo, DMODEL, DMODEL, DMODEL);
        wtrans_kernel<<<dim3(DMODEL / 32, DMODEL / 32), blk, 0, stream>>>(Wk + qo, wtk + qo, DMODEL, DMODEL, DMODEL);
        wtrans_kernel<<<dim3(DMODEL / 32, DMODEL / 32), blk, 0, stream>>>(Wv + qo, wtv + qo, DMODEL, DMODEL, DMODEL);
        wtrans_kernel<<<dim3(DMODEL / 32, DMODEL / 32), blk, 0, stream>>>(Wo + qo, wto + qo, DMODEL, DMODEL, DMODEL);
        wtrans_kernel<<<dim3(DMODEL / 32, DFF / 32), blk, 0, stream>>>(Wf1 + (size_t)i * DMODEL * DFF,
                                                                       wtf1 + (size_t)i * DFF * DMODEL, DMODEL, DFF, DMODEL);
        wtrans_kernel<<<dim3(DFF / 32, DMODEL / 32), blk, 0, stream>>>(Wf2 + (size_t)i * DFF * DMODEL,
                                                                       wtf2 + (size_t)i * DMODEL * DFF, DFF, DMODEL, DFF);
    }

    // ---- motion encoder ----
    concat_kernel<<<(BATCH * TSEQ * KPAD + 255) / 256, blk, 0, stream>>>(x, mask, traj, acat);
    {
        dim3 g(DMODEL / BN, (BATCH * TSEQ + BM - 1) / BM);
        gemm_kernel<1, false, true ><<<g, blk, 0, stream>>>(acat, wtme1, me_b1, me_a1, nullptr, me1b,
                                                            BATCH * TSEQ, DMODEL, KPAD);
        gemm_kernel<1, false, false><<<g, blk, 0, stream>>>(me1b, wtme2, me_b2, me_a2, qf, nullptr,
                                                            BATCH * TSEQ, DMODEL, DMODEL);
    }
    assemble_kernel<<<(BATCH * T2LEN * DMODEL + 255) / 256, blk, 0, stream>>>(qf, mu_tok, lv_tok, h);
    relpos_kernel<<<(WINSZ * DHEAD + 255) / 256, blk, 0, stream>>>(rp_W1, rp_b1, rp_a, rp_W2, rp_b2, tb);

    dim3 gP(DMODEL / BN, (MROWS + BM - 1) / BM);   // 4 x 65
    dim3 gF(DFF / BN,    (MROWS + BM - 1) / BM);   // 16 x 65
    for (int i = 0; i < NLAYERS; ++i) {
        const size_t qo = (size_t)i * DMODEL * DMODEL;

        ln_kernel<<<MROWS / 8, blk, 0, stream>>>(h, ln1_g + i * DMODEL, ln1_b + i * DMODEL, hnb, MROWS);
        gemm_kernel<0, false, false><<<gP, blk, 0, stream>>>(hnb, wtq + qo, bq + i * DMODEL, nullptr, qf, nullptr, MROWS, DMODEL, DMODEL);
        gemm_kernel<0, false, false><<<gP, blk, 0, stream>>>(hnb, wtk + qo, bk + i * DMODEL, nullptr, kf, nullptr, MROWS, DMODEL, DMODEL);
        gemm_kernel<0, false, false><<<gP, blk, 0, stream>>>(hnb, wtv + qo, bv + i * DMODEL, nullptr, vf, nullptr, MROWS, DMODEL, DMODEL);
        attn_kernel<<<(BATCH * T2LEN * NHEADS) / 8, blk, 0, stream>>>(qf, kf, vf, tb, attb);
        gemm_kernel<0, true, false ><<<gP, blk, 0, stream>>>(attb, wto + qo, bo + i * DMODEL, nullptr, h, nullptr, MROWS, DMODEL, DMODEL);
        ln_kernel<<<MROWS / 8, blk, 0, stream>>>(h, ln2_g + i * DMODEL, ln2_b + i * DMODEL, hnb, MROWS);
        gemm_kernel<1, false, true ><<<gF, blk, 0, stream>>>(hnb, wtf1 + (size_t)i * DFF * DMODEL, bf1 + i * DFF, af + i,
                                                             nullptr, ff1b, MROWS, DFF, DMODEL);
        gemm_kernel<0, true, false ><<<gP, blk, 0, stream>>>(ff1b, wtf2 + (size_t)i * DMODEL * DFF, bf2 + i * DMODEL, nullptr,
                                                             h, nullptr, MROWS, DMODEL, DFF);
    }

    out_kernel<<<(2 * BATCH * DMODEL + 255) / 256, blk, 0, stream>>>(h, outf);
}
